// MultiHeadAttention_17008070492182
// MI455X (gfx1250) — matrix operand order repaired and checked by reading, not compiled, not run on hardware
//
#include <hip/hip_runtime.h>
#include <hip/hip_bf16.h>

// ---------------------------------------------------------------------------
// MI455X / gfx1250 multi-head attention, bf16 WMMA pipeline.
//   DIM=1024, HEADS=16, HEAD_DIM=64, B=4, N=2048, M=B*N=8192
// ---------------------------------------------------------------------------

#define DIM      1024
#define HEADS    16
#define HD       64
#define BATCH    4
#define SEQ      2048
#define MROWS    (BATCH * SEQ)   // 8192

typedef __attribute__((ext_vector_type(16))) __bf16 v16bf;
typedef __attribute__((ext_vector_type(8)))  __bf16 v8bf;
typedef __attribute__((ext_vector_type(8)))  float  v8f;
typedef int v4i __attribute__((vector_size(16)));

#if __has_builtin(__builtin_amdgcn_global_load_async_to_lds_b128)
#define HAVE_ASYNC_LDS 1
typedef __attribute__((address_space(1))) v4i* gv4i_p;
typedef __attribute__((address_space(3))) v4i* lv4i_p;
#else
#define HAVE_ASYNC_LDS 0
#endif

#if __has_builtin(__builtin_amdgcn_permlane16)
#define HAVE_PERMLANE16 1
#else
#define HAVE_PERMLANE16 0
#endif

union FragA {
    v16bf v;
    v8bf  h[2];
};

// round-to-nearest-even f32 -> bf16, bit level
__device__ __forceinline__ __bf16 f2bf(float f) {
    union { float f; unsigned u; } uf; uf.f = f;
    unsigned r = uf.u + 0x7FFFu + ((uf.u >> 16) & 1u);
    unsigned short hs = (unsigned short)(r >> 16);
    union { unsigned short s; __bf16 b; } ub; ub.s = hs;
    return ub.b;
}

// xor-butterfly within 16-lane rows: VALU permlane16 (no LDS pipe traffic),
// falls back to __shfl_xor (ds_bpermute) if the builtin is missing.
__device__ __forceinline__ float bfly16(float x, unsigned s0, unsigned s1, int xm) {
#if HAVE_PERMLANE16
    union { float f; int i; } u; u.f = x;
    u.i = __builtin_amdgcn_permlane16(u.i, u.i, s0, s1, false, false);
    return u.f;
#else
    return __shfl_xor(x, xm, 16);
#endif
}
__device__ __forceinline__ float rowmax16(float x) {
    x = fmaxf(x, bfly16(x, 0x67452301u, 0xEFCDAB89u, 1));  // lane ^ 1
    x = fmaxf(x, bfly16(x, 0x54761032u, 0xDCFE98BAu, 2));  // lane ^ 2
    x = fmaxf(x, bfly16(x, 0x32107654u, 0xBA98FEDCu, 4));  // lane ^ 4
    x = fmaxf(x, bfly16(x, 0xFEDCBA98u, 0x76543210u, 8));  // lane ^ 8
    return x;
}
__device__ __forceinline__ float rowsum16(float x) {
    x += bfly16(x, 0x67452301u, 0xEFCDAB89u, 1);
    x += bfly16(x, 0x54761032u, 0xDCFE98BAu, 2);
    x += bfly16(x, 0x32107654u, 0xBA98FEDCu, 4);
    x += bfly16(x, 0xFEDCBA98u, 0x76543210u, 8);
    return x;
}

__device__ __forceinline__ void wait_async_lds() {
#if HAVE_ASYNC_LDS
#if __has_builtin(__builtin_amdgcn_s_wait_asynccnt)
    __builtin_amdgcn_s_wait_asynccnt(0);
#else
    asm volatile("s_wait_asynccnt 0x0" ::: "memory");
#endif
#endif
}

// ---------------------------------------------------------------------------
// f32 -> bf16 conversion, 8 elems / thread
// ---------------------------------------------------------------------------
__global__ __launch_bounds__(256) void cvt_f32_bf16(const float* __restrict__ s,
                                                    __bf16* __restrict__ d, int n) {
    int i = (blockIdx.x * 256 + threadIdx.x) * 8;
    if (i + 7 >= n) return;
    float4 f0 = *(const float4*)(s + i);
    float4 f1 = *(const float4*)(s + i + 4);
    v8bf o;
    o[0] = f2bf(f0.x); o[1] = f2bf(f0.y); o[2] = f2bf(f0.z); o[3] = f2bf(f0.w);
    o[4] = f2bf(f1.x); o[5] = f2bf(f1.y); o[6] = f2bf(f1.z); o[7] = f2bf(f1.w);
    *(v8bf*)(d + i) = o;
}

// ---------------------------------------------------------------------------
// C = A(MxK bf16 rm) @ W(NxK bf16 rm)^T + bias
// 8 waves as 2(M) x 4(N); wave tile = 32M x 64N (8 accs); block = 64M x 256N.
// Two-stage register pipeline over K (step 32 per stage, 64 per loop).
// ---------------------------------------------------------------------------
struct KFrags { FragA a0, a1; v16bf b0, b1, b2, b3; };

__device__ __forceinline__ void load_frags(
    KFrags& f, const __bf16* Ar0, const __bf16* Ar1,
    const __bf16* B0, const __bf16* B1, const __bf16* B2, const __bf16* B3,
    int kk, int lg)
{
    f.a0.h[0] = *(const v8bf*)(Ar0 + kk + lg * 16);
    f.a0.h[1] = *(const v8bf*)(Ar0 + kk + lg * 16 + 8);
    f.a1.h[0] = *(const v8bf*)(Ar1 + kk + lg * 16);
    f.a1.h[1] = *(const v8bf*)(Ar1 + kk + lg * 16 + 8);
    f.b0 = *(const v16bf*)(B0 + kk);
    f.b1 = *(const v16bf*)(B1 + kk);
    f.b2 = *(const v16bf*)(B2 + kk);
    f.b3 = *(const v16bf*)(B3 + kk);
}

__device__ __forceinline__ void mma_step(const KFrags& f, v8f acc[2][4]) {
    acc[0][0] = __builtin_amdgcn_wmma_f32_16x16x32_bf16(false, f.a0.v, false, f.b0, (short)0, acc[0][0], false, false);
    acc[0][1] = __builtin_amdgcn_wmma_f32_16x16x32_bf16(false, f.a0.v, false, f.b1, (short)0, acc[0][1], false, false);
    acc[0][2] = __builtin_amdgcn_wmma_f32_16x16x32_bf16(false, f.a0.v, false, f.b2, (short)0, acc[0][2], false, false);
    acc[0][3] = __builtin_amdgcn_wmma_f32_16x16x32_bf16(false, f.a0.v, false, f.b3, (short)0, acc[0][3], false, false);
    acc[1][0] = __builtin_amdgcn_wmma_f32_16x16x32_bf16(false, f.a1.v, false, f.b0, (short)0, acc[1][0], false, false);
    acc[1][1] = __builtin_amdgcn_wmma_f32_16x16x32_bf16(false, f.a1.v, false, f.b1, (short)0, acc[1][1], false, false);
    acc[1][2] = __builtin_amdgcn_wmma_f32_16x16x32_bf16(false, f.a1.v, false, f.b2, (short)0, acc[1][2], false, false);
    acc[1][3] = __builtin_amdgcn_wmma_f32_16x16x32_bf16(false, f.a1.v, false, f.b3, (short)0, acc[1][3], false, false);
}

template <bool OUT_BF16>
__global__ __launch_bounds__(256) void gemm_bf16_wmma(
    const __bf16* __restrict__ A, const __bf16* __restrict__ W,
    const float* __restrict__ bias, void* __restrict__ Cout,
    int M, int N, int K)
{
    const int lane = threadIdx.x & 31;
    const int wave = threadIdx.x >> 5;
    const int lm = lane & 15;
    const int lg = lane >> 4;
    const int wm = wave & 1;
    const int wn = wave >> 1;
    const int m0 = blockIdx.x * 64 + wm * 32;
    const int n0 = blockIdx.y * 256 + wn * 64;

    const __bf16* Ar0 = A + (size_t)(m0 + lm) * K;
    const __bf16* Ar1 = A + (size_t)(m0 + 16 + lm) * K;
    const __bf16* B0 = W + (size_t)(n0 + 0 * 16 + lm) * K + (size_t)lg * 16;
    const __bf16* B1 = W + (size_t)(n0 + 1 * 16 + lm) * K + (size_t)lg * 16;
    const __bf16* B2 = W + (size_t)(n0 + 2 * 16 + lm) * K + (size_t)lg * 16;
    const __bf16* B3 = W + (size_t)(n0 + 3 * 16 + lm) * K + (size_t)lg * 16;

    v8f acc[2][4] = {};
    KFrags f0, f1;
    load_frags(f0, Ar0, Ar1, B0, B1, B2, B3, 0, lg);

    int kk = 0;
    for (; kk + 64 < K; kk += 64) {
        load_frags(f1, Ar0, Ar1, B0, B1, B2, B3, kk + 32, lg);
        __builtin_prefetch((const void*)(Ar0 + kk + 512), 0, 0);
        __builtin_prefetch((const void*)(Ar1 + kk + 512), 0, 0);
        mma_step(f0, acc);
        load_frags(f0, Ar0, Ar1, B0, B1, B2, B3, kk + 64, lg);
        mma_step(f1, acc);
    }
    load_frags(f1, Ar0, Ar1, B0, B1, B2, B3, kk + 32, lg);
    mma_step(f0, acc);
    mma_step(f1, acc);

    float bj[4];
#pragma unroll
    for (int j = 0; j < 4; ++j) bj[j] = bias[n0 + j * 16 + lm];

#pragma unroll
    for (int i = 0; i < 2; ++i)
#pragma unroll
        for (int j = 0; j < 4; ++j)
#pragma unroll
            for (int r = 0; r < 8; ++r) {
                size_t row = (size_t)(m0 + i * 16 + r + 8 * lg);
                size_t idx = row * (size_t)N + (size_t)(n0 + j * 16 + lm);
                float val = acc[i][j][r] + bj[j];
                if constexpr (OUT_BF16) ((__bf16*)Cout)[idx] = f2bf(val);
                else                    ((float*)Cout)[idx] = val;
            }
}

// ---------------------------------------------------------------------------
// Flash attention: grid = (SEQ/128, HEADS, BATCH), block = 256 (8 waves).
// Waves cooperatively stage 32-key K/V blocks in LDS; K via async global->LDS
// copies (ASYNCcnt) when available, V transposed through registers.
// ---------------------------------------------------------------------------
__global__ __launch_bounds__(256) void flash_attn_wmma(
    const __bf16* __restrict__ Qg, const __bf16* __restrict__ Kg,
    const __bf16* __restrict__ Vg, __bf16* __restrict__ Og)
{
    __shared__ __bf16 Kb[32 * 64];      // [key][hd]
    __shared__ __bf16 Vt[64 * 32];      // [hd][key]   (transposed)
    __shared__ __bf16 Pls[8 * 16 * 32]; // per-wave P tile [row][key]

    const int lane = threadIdx.x & 31;
    const int wave = threadIdx.x >> 5;
    const int lm = lane & 15;
    const int lg = lane >> 4;
    const int h  = blockIdx.y;
    const int bz = blockIdx.z;
    const int qrow0 = blockIdx.x * 128 + wave * 16;
    const float scale = 0.125f;  // 1/sqrt(64)

    __bf16* Pw = &Pls[wave * 16 * 32];

    FragA qf[2];
    {
        const __bf16* Qrow = Qg + ((size_t)(bz * SEQ + qrow0 + lm)) * DIM + h * HD;
#pragma unroll
        for (int ks = 0; ks < 2; ++ks) {
            qf[ks].h[0] = *(const v8bf*)(Qrow + ks * 32 + lg * 16);
            qf[ks].h[1] = *(const v8bf*)(Qrow + ks * 32 + lg * 16 + 8);
        }
    }

    v8f o[4] = {};
    float Mr[8], Lr[8];
#pragma unroll
    for (int r = 0; r < 8; ++r) { Mr[r] = -3.402823466e38f; Lr[r] = 0.0f; }

    const int tid  = threadIdx.x;
    const int skey = tid >> 3;         // 0..31
    const int sdof = (tid & 7) * 8;    // 0..56

    for (int kb = 0; kb < SEQ; kb += 32) {
        __syncthreads();
        // ---- stage K (async global->LDS if available) and V (transposed) ----
        {
            const __bf16* Ksrc = Kg + ((size_t)(bz * SEQ + kb + skey)) * DIM + h * HD + sdof;
#if HAVE_ASYNC_LDS
            __builtin_amdgcn_global_load_async_to_lds_b128(
                (gv4i_p)(Ksrc),
                (lv4i_p)(&Kb[skey * 64 + sdof]),
                0, 0);
#else
            *(v8bf*)(&Kb[skey * 64 + sdof]) = *(const v8bf*)Ksrc;
#endif
            const __bf16* Vsrc = Vg + ((size_t)(bz * SEQ + kb + skey)) * DIM + h * HD + sdof;
            v8bf vv = *(const v8bf*)Vsrc;
#pragma unroll
            for (int i = 0; i < 8; ++i) Vt[(sdof + i) * 32 + skey] = vv[i];
        }
        wait_async_lds();
        __syncthreads();

        // ---- S = Q @ K^T  (two 16-key tiles) ----
        v8f s0 = {}, s1 = {};
#pragma unroll
        for (int ks = 0; ks < 2; ++ks) {
            v16bf k0 = *(const v16bf*)(&Kb[(0 * 16 + lm) * 64 + ks * 32 + lg * 16]);
            v16bf k1 = *(const v16bf*)(&Kb[(1 * 16 + lm) * 64 + ks * 32 + lg * 16]);
            s0 = __builtin_amdgcn_wmma_f32_16x16x32_bf16(false, qf[ks].v, false, k0, (short)0, s0, false, false);
            s1 = __builtin_amdgcn_wmma_f32_16x16x32_bf16(false, qf[ks].v, false, k1, (short)0, s1, false, false);
        }

        // ---- online softmax update; P -> bf16 into LDS ----
        float alpha[8];
#pragma unroll
        for (int r = 0; r < 8; ++r) {
            float x0 = s0[r] * scale;
            float x1 = s1[r] * scale;
            float mx = rowmax16(fmaxf(x0, x1));
            float newM = fmaxf(Mr[r], mx);
            alpha[r] = __expf(Mr[r] - newM);
            Mr[r] = newM;
            float p0 = __expf(x0 - newM);
            float p1 = __expf(x1 - newM);
            float rs = rowsum16(p0 + p1);
            Lr[r] = Lr[r] * alpha[r] + rs;
            Pw[(r + 8 * lg) * 32 + lm]      = f2bf(p0);
            Pw[(r + 8 * lg) * 32 + 16 + lm] = f2bf(p1);
        }

#pragma unroll
        for (int j = 0; j < 4; ++j)
#pragma unroll
            for (int r = 0; r < 8; ++r) o[j][r] *= alpha[r];

        // ---- P fragment (A-layout) from LDS; wave-local so no barrier ----
        FragA pf;
        pf.h[0] = *(const v8bf*)(&Pw[lm * 32 + lg * 16]);
        pf.h[1] = *(const v8bf*)(&Pw[lm * 32 + lg * 16 + 8]);

        // ---- O += P @ V  (4 d-tiles of 16) ----
#pragma unroll
        for (int j = 0; j < 4; ++j) {
            v16bf vf = *(const v16bf*)(&Vt[(j * 16 + lm) * 32 + lg * 16]);
            o[j] = __builtin_amdgcn_wmma_f32_16x16x32_bf16(false, pf.v, false, vf, (short)0, o[j], false, false);
        }
    }

    // ---- normalize and store bf16 output ----
#pragma unroll
    for (int j = 0; j < 4; ++j)
#pragma unroll
        for (int r = 0; r < 8; ++r) {
            float val = o[j][r] / Lr[r];
            size_t idx = ((size_t)(bz * SEQ + qrow0 + r + 8 * lg)) * DIM + h * HD + j * 16 + lm;
            Og[idx] = f2bf(val);
        }
}

// ---------------------------------------------------------------------------
// host-side orchestration
// ---------------------------------------------------------------------------
extern "C" void kernel_launch(void* const* d_in, const int* in_sizes, int n_in,
                              void* d_out, int out_size, void* d_ws, size_t ws_size,
                              hipStream_t stream) {
    const float* x  = (const float*)d_in[0];
    const float* Wq = (const float*)d_in[1];
    const float* bq = (const float*)d_in[2];
    const float* Wk = (const float*)d_in[3];
    const float* bk = (const float*)d_in[4];
    const float* Wv = (const float*)d_in[5];
    const float* bv = (const float*)d_in[6];
    const float* Wo = (const float*)d_in[7];
    const float* bo = (const float*)d_in[8];

    char* ws = (char*)d_ws;
    const size_t xN = (size_t)MROWS * DIM;     // 8,388,608
    const size_t wN = (size_t)DIM * DIM;       // 1,048,576

    __bf16* xbf  = (__bf16*)ws;                      ws += xN * 2;
    __bf16* Wqb  = (__bf16*)ws;                      ws += wN * 2;
    __bf16* Wkb  = (__bf16*)ws;                      ws += wN * 2;
    __bf16* Wvb  = (__bf16*)ws;                      ws += wN * 2;
    __bf16* Wob  = (__bf16*)ws;                      ws += wN * 2;
    __bf16* qbf  = (__bf16*)ws;                      ws += xN * 2;
    __bf16* kbf  = (__bf16*)ws;                      ws += xN * 2;
    __bf16* vbf  = (__bf16*)ws;                      ws += xN * 2;
    __bf16* obf  = (__bf16*)ws;                      ws += xN * 2;

    cvt_f32_bf16<<<(int)(xN / 2048), 256, 0, stream>>>(x,  xbf, (int)xN);
    cvt_f32_bf16<<<(int)(wN / 2048), 256, 0, stream>>>(Wq, Wqb, (int)wN);
    cvt_f32_bf16<<<(int)(wN / 2048), 256, 0, stream>>>(Wk, Wkb, (int)wN);
    cvt_f32_bf16<<<(int)(wN / 2048), 256, 0, stream>>>(Wv, Wvb, (int)wN);
    cvt_f32_bf16<<<(int)(wN / 2048), 256, 0, stream>>>(Wo, Wob, (int)wN);

    dim3 ggrid(MROWS / 64, DIM / 256);
    gemm_bf16_wmma<true><<<ggrid, 256, 0, stream>>>(xbf, Wqb, bq, (void*)qbf, MROWS, DIM, DIM);
    gemm_bf16_wmma<true><<<ggrid, 256, 0, stream>>>(xbf, Wkb, bk, (void*)kbf, MROWS, DIM, DIM);
    gemm_bf16_wmma<true><<<ggrid, 256, 0, stream>>>(xbf, Wvb, bv, (void*)vbf, MROWS, DIM, DIM);

    dim3 agrid(SEQ / 128, HEADS, BATCH);
    flash_attn_wmma<<<agrid, 256, 0, stream>>>(qbf, kbf, vbf, obf);

    gemm_bf16_wmma<false><<<ggrid, 256, 0, stream>>>(obf, Wob, bo, d_out, MROWS, DIM, DIM);
}
